// GPT2Attention_60301340836235
// MI455X (gfx1250) — matrix operand order repaired and checked by reading, not compiled, not run on hardware
//
#include <hip/hip_runtime.h>
#include <hip/hip_bf16.h>

typedef __bf16 bf16_t;
typedef bf16_t v16bf __attribute__((ext_vector_type(16)));
typedef bf16_t v8bf  __attribute__((ext_vector_type(8)));
typedef float  v8f   __attribute__((ext_vector_type(8)));
typedef __attribute__((address_space(3))) char lds_char;

#define N_HEAD 12
#define HEAD   64
#define HIDDEN 768
#define SEQ    2048
#define BATCH  2
#define ROWS   (BATCH * SEQ)   // 4096

// ---------- helpers ----------
__device__ __forceinline__ bf16_t f2bf(float f) {
  unsigned u = __float_as_uint(f);
  u += 0x7FFFu + ((u >> 16) & 1u);                 // round-to-nearest-even
  unsigned short s = (unsigned short)(u >> 16);
  return __builtin_bit_cast(bf16_t, s);
}

__device__ __forceinline__ v8f wmma_bf16(v16bf a, v16bf b, v8f c) {
  // D = A(16x32 bf16) * B(32x16 bf16) + C(16x16 f32)
  return __builtin_amdgcn_wmma_f32_16x16x32_bf16(false, a, false, b, (short)0, c,
                                                 false, false);
}

// A fragment 16x32 from row-major [m, k] source (lda in elements), tile at (row0, kb)
__device__ __forceinline__ v16bf load_afrag(const bf16_t* base, int lda, int kb, int lane) {
  int r = lane & 15, half = lane >> 4;
  const bf16_t* p = base + (size_t)r * lda + kb + 8 * half;
  v8bf c0 = *(const v8bf*)(p);
  v8bf c1 = *(const v8bf*)(p + 16);
  return __builtin_shufflevector(c0, c1, 0,1,2,3,4,5,6,7,8,9,10,11,12,13,14,15);
}

// B fragment 32x16 from N-major source Bt[n][k] (ldb in elements); base points at row n0
__device__ __forceinline__ v16bf load_bfrag(const bf16_t* base, int ldb, int kb, int lane) {
  int n = lane & 15, half = lane >> 4;
  const bf16_t* p = base + (size_t)n * ldb + kb + 8 * half;
  v8bf c0 = *(const v8bf*)(p);
  v8bf c1 = *(const v8bf*)(p + 16);
  return __builtin_shufflevector(c0, c1, 0,1,2,3,4,5,6,7,8,9,10,11,12,13,14,15);
}

__device__ __forceinline__ float redmax16(float v) {
  v = fmaxf(v, __shfl_xor(v, 1, 32));
  v = fmaxf(v, __shfl_xor(v, 2, 32));
  v = fmaxf(v, __shfl_xor(v, 4, 32));
  v = fmaxf(v, __shfl_xor(v, 8, 32));
  return v;
}
__device__ __forceinline__ float redsum16(float v) {
  v += __shfl_xor(v, 1, 32);
  v += __shfl_xor(v, 2, 32);
  v += __shfl_xor(v, 4, 32);
  v += __shfl_xor(v, 8, 32);
  return v;
}

// Async-copy `bytes` (multiple of 16*blockDim) from contiguous global to LDS.
// Uses CDNA5 GLOBAL_LOAD_ASYNC_TO_LDS_B128 (ASYNCcnt-tracked), then waits + barrier.
__device__ __forceinline__ void async_panel_copy(void* lds_dst, const void* gsrc, int bytes) {
  lds_char* l3 = (lds_char*)lds_dst;
  const char* g = (const char*)gsrc;
  int chunks = bytes >> 4;
  for (int i = threadIdx.x; i < chunks; i += blockDim.x) {
    unsigned loff = (unsigned)(size_t)(l3 + i * 16);
    unsigned long long ga = (unsigned long long)(size_t)(g + i * 16);
    asm volatile("global_load_async_to_lds_b128 %0, %1, off"
                 :: "v"(loff), "v"(ga) : "memory");
  }
  asm volatile("s_wait_asynccnt 0" ::: "memory");
  __syncthreads();
}

// ---------- conversion / transpose kernels ----------
__global__ void cvt_bf16_kernel(const float* __restrict__ in, bf16_t* __restrict__ out, int n) {
  for (int i = blockIdx.x * blockDim.x + threadIdx.x; i < n; i += gridDim.x * blockDim.x)
    out[i] = f2bf(in[i]);
}

// Wt[n*K + k] = W[k*N + n]  (W is K x N row-major)
__global__ void transpose_bf16_kernel(const float* __restrict__ w, bf16_t* __restrict__ wt,
                                      int K, int N) {
  int total = K * N;
  for (int i = blockIdx.x * blockDim.x + threadIdx.x; i < total; i += gridDim.x * blockDim.x) {
    int k = i / N, n = i % N;
    wt[(size_t)n * K + k] = f2bf(w[i]);
  }
}

// ---------- QKV projection: qkv = X @ Wqkv + b, scattered to Q/K/Vt bf16 ----------
// grid (9, 256), block 128 (4 waves; block = 16 rows x 256 cols; wave = 16 x 64)
__global__ void qkv_gemm_kernel(const bf16_t* __restrict__ X,   // [4096, 768]
                                const bf16_t* __restrict__ Wt,  // [2304, 768] (N-major)
                                const float* __restrict__ bias, // [2304]
                                bf16_t* __restrict__ qb,        // [B,H,S,64]
                                bf16_t* __restrict__ kb,        // [B,H,S,64]
                                bf16_t* __restrict__ vt) {      // [B,H,64,S]
  __shared__ __align__(16) bf16_t apanel[16 * HIDDEN];          // 24 KB
  int lane = threadIdx.x & 31, wave = threadIdx.x >> 5;
  int mbase = blockIdx.y * 16;
  int nb0 = blockIdx.x * 256 + wave * 64;
  int ln = lane & 15, half = lane >> 4;

  async_panel_copy(apanel, X + (size_t)mbase * HIDDEN, 16 * HIDDEN * 2);

  const bf16_t* brow0 = Wt + (size_t)nb0 * HIDDEN;
  v8f acc[4] = {};
  for (int kk = 0; kk < HIDDEN; kk += 32) {
    v16bf a = load_afrag(apanel, HIDDEN, kk, lane);
#pragma unroll
    for (int t = 0; t < 4; ++t) {
      if (kk + 32 < HIDDEN)
        __builtin_prefetch(brow0 + (size_t)(16 * t + ln) * HIDDEN + kk + 32, 0, 3);
      v16bf b = load_bfrag(brow0 + (size_t)(16 * t) * HIDDEN, HIDDEN, kk, lane);
      acc[t] = wmma_bf16(a, b, acc[t]);
    }
  }
#pragma unroll
  for (int t = 0; t < 4; ++t) {
    int j = nb0 + 16 * t + ln;
    float bs = bias[j];
    int sect = j / HIDDEN;       // 0=Q 1=K 2=V
    int jj = j % HIDDEN;
    int h = jj >> 6, d = jj & 63;
#pragma unroll
    for (int r = 0; r < 8; ++r) {
      int sg = mbase + r + 8 * half;          // global row in [0,4096)
      int bb = sg >> 11, s = sg & (SEQ - 1);
      bf16_t v = f2bf(acc[t][r] + bs);
      size_t bh = (size_t)(bb * N_HEAD + h);
      if (sect == 0)      qb[(bh * SEQ + s) * HEAD + d] = v;
      else if (sect == 1) kb[(bh * SEQ + s) * HEAD + d] = v;
      else                vt[(bh * HEAD + d) * SEQ + s] = v;
    }
  }
}

// ---------- flash attention: one wave per 16-row q tile ----------
// grid 3072, block 32
__global__ void attn_kernel(const bf16_t* __restrict__ qbf,
                            const bf16_t* __restrict__ kbf,
                            const bf16_t* __restrict__ vt,
                            bf16_t* __restrict__ ctx) {   // [4096, 768] bf16
  __shared__ __align__(16) bf16_t pbuf[16 * 32];
  int lane = threadIdx.x & 31;
  int tile = blockIdx.x;
  int qt = tile & 127;              // 128 q-tiles per (b,h)
  int bh = tile >> 7;               // 0..23
  int b = bh / N_HEAD, h = bh % N_HEAD;
  int qbase = qt * 16;
  int ln = lane & 15, half = lane >> 4;

  const bf16_t* qh = qbf + (size_t)bh * SEQ * HEAD;
  const bf16_t* kh = kbf + (size_t)bh * SEQ * HEAD;
  const bf16_t* vh = vt + (size_t)bh * HEAD * SEQ;

  v16bf aq0 = load_afrag(qh + (size_t)qbase * HEAD, HEAD, 0, lane);
  v16bf aq1 = load_afrag(qh + (size_t)qbase * HEAD, HEAD, 32, lane);

  float m_run[8], l_run[8];
  v8f accO[4];
#pragma unroll
  for (int r = 0; r < 8; ++r) { m_run[r] = -1e30f; l_run[r] = 0.f; }
#pragma unroll
  for (int f = 0; f < 4; ++f) accO[f] = (v8f){};

  const float scale = 0.125f;   // 1/sqrt(64)
  int nkt = qt / 2 + 1;         // 32-key groups needed (causal)
  for (int it = 0; it < nkt; ++it) {
    int kv = it * 32;
    v8f s0 = {}, s1 = {};
    {
      v16bf bk = load_bfrag(kh + (size_t)kv * HEAD, HEAD, 0, lane);
      s0 = wmma_bf16(aq0, bk, s0);
      bk = load_bfrag(kh + (size_t)kv * HEAD, HEAD, 32, lane);
      s0 = wmma_bf16(aq1, bk, s0);
      bk = load_bfrag(kh + (size_t)(kv + 16) * HEAD, HEAD, 0, lane);
      s1 = wmma_bf16(aq0, bk, s1);
      bk = load_bfrag(kh + (size_t)(kv + 16) * HEAD, HEAD, 32, lane);
      s1 = wmma_bf16(aq1, bk, s1);
    }
    float p0[8], p1[8];
#pragma unroll
    for (int r = 0; r < 8; ++r) {
      int qg = qbase + r + 8 * half;           // global query row this lane owns
      float v0 = s0[r] * scale; if (kv + ln > qg)      v0 = -1e30f;
      float v1 = s1[r] * scale; if (kv + 16 + ln > qg) v1 = -1e30f;
      // v0/v1 are different columns of the SAME row -> one fused reduction each
      float rm = redmax16(fmaxf(v0, v1));
      float mnew = fmaxf(m_run[r], rm);
      float sc = __expf(m_run[r] - mnew);
      float e0 = __expf(v0 - mnew);
      float e1 = __expf(v1 - mnew);
      l_run[r] = l_run[r] * sc + redsum16(e0 + e1);
      m_run[r] = mnew;
      p0[r] = e0; p1[r] = e1;
      accO[0][r] *= sc; accO[1][r] *= sc; accO[2][r] *= sc; accO[3][r] *= sc;
    }
    // stage P (C layout -> row-major LDS [16 rows][32 keys])
#pragma unroll
    for (int r = 0; r < 8; ++r) {
      int m = r + 8 * half;
      pbuf[m * 32 + ln]      = f2bf(p0[r]);
      pbuf[m * 32 + 16 + ln] = f2bf(p1[r]);
    }
    __syncthreads();   // single-wave WG: barrier is s_nop, LDS wait retained
    v16bf pa = load_afrag(pbuf, 32, 0, lane);
#pragma unroll
    for (int f = 0; f < 4; ++f) {
      v16bf bv = load_bfrag(vh + (size_t)(16 * f) * SEQ, SEQ, kv, lane);
      accO[f] = wmma_bf16(pa, bv, accO[f]);
    }
    __syncthreads();   // protect pbuf before next iteration's writes
  }
#pragma unroll
  for (int r = 0; r < 8; ++r) {
    float inv = 1.0f / l_run[r];
    int sg = qbase + r + 8 * half;
    size_t rowoff = ((size_t)(b * SEQ + sg)) * HIDDEN + h * HEAD;
#pragma unroll
    for (int f = 0; f < 4; ++f)
      ctx[rowoff + 16 * f + ln] = f2bf(accO[f][r] * inv);
  }
}

// ---------- output projection: out = ctx @ Wproj + b (fp32 out) ----------
// grid (3, 256), block 128 (4 waves; wave = 16 x 64)
__global__ void proj_gemm_kernel(const bf16_t* __restrict__ C,  // [4096, 768]
                                 const bf16_t* __restrict__ Pt, // [768, 768] N-major
                                 const float* __restrict__ bias,
                                 float* __restrict__ out) {
  __shared__ __align__(16) bf16_t apanel[16 * HIDDEN];          // 24 KB
  int lane = threadIdx.x & 31, wave = threadIdx.x >> 5;
  int mbase = blockIdx.y * 16;
  int nb0 = blockIdx.x * 256 + wave * 64;
  int ln = lane & 15, half = lane >> 4;

  async_panel_copy(apanel, C + (size_t)mbase * HIDDEN, 16 * HIDDEN * 2);

  const bf16_t* brow0 = Pt + (size_t)nb0 * HIDDEN;
  v8f acc[4] = {};
  for (int kk = 0; kk < HIDDEN; kk += 32) {
    v16bf a = load_afrag(apanel, HIDDEN, kk, lane);
#pragma unroll
    for (int t = 0; t < 4; ++t) {
      if (kk + 32 < HIDDEN)
        __builtin_prefetch(brow0 + (size_t)(16 * t + ln) * HIDDEN + kk + 32, 0, 3);
      v16bf b = load_bfrag(brow0 + (size_t)(16 * t) * HIDDEN, HIDDEN, kk, lane);
      acc[t] = wmma_bf16(a, b, acc[t]);
    }
  }
#pragma unroll
  for (int t = 0; t < 4; ++t) {
    float bs = bias[nb0 + 16 * t + ln];
#pragma unroll
    for (int r = 0; r < 8; ++r) {
      int m = mbase + r + 8 * half;
      out[(size_t)m * HIDDEN + nb0 + 16 * t + ln] = acc[t][r] + bs;
    }
  }
}

// ---------- launch ----------
extern "C" void kernel_launch(void* const* d_in, const int* in_sizes, int n_in,
                              void* d_out, int out_size, void* d_ws, size_t ws_size,
                              hipStream_t stream) {
  const float* hs    = (const float*)d_in[0];  // [2,2048,768]
  const float* wqkv  = (const float*)d_in[1];  // [768,2304]
  const float* bqkv  = (const float*)d_in[2];  // [2304]
  const float* wproj = (const float*)d_in[3];  // [768,768]
  const float* bproj = (const float*)d_in[4];  // [768]
  float* out = (float*)d_out;

  char* w = (char*)d_ws;
  bf16_t* Xbf = (bf16_t*)w;                 w += (size_t)ROWS * HIDDEN * 2;       // 6.29 MB
  bf16_t* Wqt = (bf16_t*)w;                 w += (size_t)3 * HIDDEN * HIDDEN * 2; // 3.54 MB
  bf16_t* Ppt = (bf16_t*)w;                 w += (size_t)HIDDEN * HIDDEN * 2;     // 1.18 MB
  bf16_t* Qb  = (bf16_t*)w;                 w += (size_t)ROWS * HIDDEN * 2;       // 6.29 MB
  bf16_t* Kb  = (bf16_t*)w;                 w += (size_t)ROWS * HIDDEN * 2;
  bf16_t* Vt  = (bf16_t*)w;                 w += (size_t)ROWS * HIDDEN * 2;
  bf16_t* Ctx = (bf16_t*)w;                 w += (size_t)ROWS * HIDDEN * 2;

  // 1) convert activations + transpose weights to bf16
  cvt_bf16_kernel<<<1024, 256, 0, stream>>>(hs, Xbf, ROWS * HIDDEN);
  transpose_bf16_kernel<<<1024, 256, 0, stream>>>(wqkv, Wqt, HIDDEN, 3 * HIDDEN);
  transpose_bf16_kernel<<<512, 256, 0, stream>>>(wproj, Ppt, HIDDEN, HIDDEN);

  // 2) QKV projection -> Q/K (s-major) and V (d-major transposed)
  qkv_gemm_kernel<<<dim3(9, 256), 128, 0, stream>>>(Xbf, Wqt, bqkv, Qb, Kb, Vt);

  // 3) causal flash attention, one wave per 16-query tile
  attn_kernel<<<BATCH * N_HEAD * (SEQ / 16), 32, 0, stream>>>(Qb, Kb, Vt, Ctx);

  // 4) output projection to fp32
  proj_gemm_kernel<<<dim3(3, 256), 128, 0, stream>>>(Ctx, Ppt, bproj, out);
}
